// MyRNN_53386443489406
// MI455X (gfx1250) — matrix operand order repaired and checked by reading, not compiled, not run on hardware
//
#include <hip/hip_runtime.h>

// MyRNN for MI455X (gfx1250, wave32, WMMA bf16 16x16x32, fp32 accumulate).
//
// Pipeline:
//   k_f32_to_bf16_vec4 : X fp32 -> bf16                  (parallel)
//   k_transpose_to_bf16: W_xh/W_hh/W_ho -> N-major bf16  (parallel)
//   k_gemm_xh          : xh = X @ W_xh   (WMMA, 68.7 GF, parallel)
//   k_recurrent        : serial scan; W_hh register-resident (128 VGPRs/wave),
//                        H in bank-conflict-free padded LDS, 2 WGs batch-split
//   k_gemm_out         : out = Hseq @ W_ho + B_o (WMMA, 68.7 GF, parallel)

#define BB 32
#define TT 2048
#define VV 1024
#define HH 512
#define HPAD 528   // LDS row stride: 1056 B -> 8-bank rotation/row, no conflicts

typedef __attribute__((ext_vector_type(16))) __bf16 v16bf;
typedef __attribute__((ext_vector_type(8)))  float  v8f;

__device__ __forceinline__ unsigned short f2bf(float x) {
  union { float f; unsigned u; } v; v.f = x;
  unsigned r = v.u + 0x7FFFu + ((v.u >> 16) & 1u);   // round-to-nearest-even
  return (unsigned short)(r >> 16);
}
__device__ __forceinline__ float bf2f(unsigned short b) {
  union { float f; unsigned u; } v; v.u = ((unsigned)b) << 16;
  return v.f;
}

union Frag { v16bf v; uint4 q[2]; unsigned short us[16]; };

// A fragment, 16x32 bf16 tile from row-major memory (stride in elements).
// ISA layout: lanes 0-15 hold rows m=lane, K in {0..7, 16..23};
//             lanes 16-31 hold rows m=lane-16, K in {8..15, 24..31}.
__device__ __forceinline__ v16bf load_fragA(const unsigned short* base, int stride) {
  const int lane = threadIdx.x & 31;
  const int half = lane >> 4;
  const int m    = lane & 15;
  const unsigned short* p = base + m * stride + half * 8;
  Frag f;
  f.q[0] = *(const uint4*)(p);        // K = half*8 .. half*8+7
  f.q[1] = *(const uint4*)(p + 16);   // K = 16+half*8 .. +7
  return f.v;
}

// B fragment, 32x16 bf16 tile from an N-major (pre-transposed) weight matrix.
// ISA layout: lanes 0-15 hold col n=lane, K in {0..7, 16..23}; lanes 16-31 hold K in {8..15, 24..31}.
// base points at element [n0][k0]; stride = K dimension length.
__device__ __forceinline__ v16bf load_fragB(const unsigned short* base, int stride) {
  const int lane  = threadIdx.x & 31;
  const int khalf = lane >> 4;
  const int n     = lane & 15;
  const unsigned short* p = base + n * stride + khalf * 8;
  Frag f;
  f.q[0] = *(const uint4*)(p);
  f.q[1] = *(const uint4*)(p + 16);
  return f.v;
}

__device__ __forceinline__ v8f wmma_bf16(v16bf a, v16bf b, v8f c) {
  return __builtin_amdgcn_wmma_f32_16x16x32_bf16(
      /*neg_a=*/false, a, /*neg_b=*/false, b,
      /*c_mod=*/(short)0, c, /*reuse_a=*/false, /*reuse_b=*/false);
}

// ---------------------------------------------------------------- converters
__global__ void k_f32_to_bf16_vec4(const float* __restrict__ src,
                                   unsigned short* __restrict__ dst,
                                   long long n4) {
  long long i = (long long)blockIdx.x * blockDim.x + threadIdx.x;
  const long long stride = (long long)gridDim.x * blockDim.x;
  for (; i < n4; i += stride) {
    float4 f = ((const float4*)src)[i];
    ushort4 u;
    u.x = f2bf(f.x); u.y = f2bf(f.y); u.z = f2bf(f.z); u.w = f2bf(f.w);
    ((ushort4*)dst)[i] = u;
  }
}

// src [rows][cols] fp32 -> dst [cols][rows] bf16
__global__ void k_transpose_to_bf16(const float* __restrict__ src,
                                    unsigned short* __restrict__ dst,
                                    int rows, int cols) {
  const int total = rows * cols;
  for (int d = blockIdx.x * blockDim.x + threadIdx.x; d < total;
       d += gridDim.x * blockDim.x) {
    const int c = d / rows, r = d - c * rows;
    dst[d] = f2bf(src[(size_t)r * cols + c]);
  }
}

// ------------------------------------------------- GEMM1: xh = X @ W_xh
// A = Xbf [B*T][V] row-major (row = b*T + t), B = WxhT [H][V] (N-major).
// Block: 8 waves, 128(M) x 64(N) tile; wave: 16x64 (4 accumulators), K=1024.
// Output stored time-major: xh[t][b][h] so the recurrent kernel reads it flat.
__global__ __launch_bounds__(256) void k_gemm_xh(
    const unsigned short* __restrict__ Xbf,
    const unsigned short* __restrict__ WxhT,
    float* __restrict__ xh) {
  const int wave = threadIdx.x >> 5;
  const int lane = threadIdx.x & 31;
  const int half = lane >> 4;
  const int n    = lane & 15;
  const int m0 = blockIdx.y * 128 + wave * 16;
  const int n0 = blockIdx.x * 64;

  v8f acc[4] = {v8f{}, v8f{}, v8f{}, v8f{}};
  const unsigned short* arow = Xbf + (size_t)m0 * VV;

#pragma unroll 2
  for (int k = 0; k < VV; k += 32) {
    __builtin_prefetch(arow + k + 128, 0, 0);          // global_prefetch_b8
    v16bf a = load_fragA(arow + k, VV);
#pragma unroll
    for (int j = 0; j < 4; ++j) {
      v16bf b = load_fragB(WxhT + (size_t)(n0 + 16 * j) * VV + k, VV);
      acc[j] = wmma_bf16(a, b, acc[j]);
    }
  }
#pragma unroll
  for (int j = 0; j < 4; ++j) {
    const int col = n0 + 16 * j + n;
#pragma unroll
    for (int r = 0; r < 8; ++r) {
      const int row = m0 + r + 8 * half;     // row = b*T + t
      const int bb = row >> 11;              // T = 2048
      const int tt = row & (TT - 1);
      xh[(size_t)tt * BB * HH + (size_t)bb * HH + col] = acc[j][r];
    }
  }
}

// --------------------------------- serial scan: H = tanh(xh_t + H@W_hh + B_h)
// 2 blocks (batch halves of 16 rows = one WMMA M-tile), 1024 threads = 32 waves.
// Wave w owns 16 output columns [16w, 16w+16) and holds its entire W_hh slice
// register-resident: 16 B-fragments = 128 VGPRs, loaded once, reused 2048x.
// Per step: 16 conflict-free LDS A-fragment loads + 16 WMMAs + tanh + barrier.
__global__ __launch_bounds__(1024, 1) void k_recurrent(
    const float* __restrict__ state,          // [B][H]
    const float* __restrict__ xh,             // [T][B][H]
    const unsigned short* __restrict__ WhhT,  // [H][H] N-major
    const float* __restrict__ Bh,             // [H]
    unsigned short* __restrict__ Hseq,        // [T*B][H], row = t*B + b
    float* __restrict__ Hfinal) {             // [B][H]
  __shared__ __align__(16) unsigned short Hcur[16 * HPAD];

  const int wave  = threadIdx.x >> 5;   // 0..31
  const int lane  = threadIdx.x & 31;
  const int half  = lane >> 4;
  const int n     = lane & 15;
  const int bbase = blockIdx.x * 16;
  const int n0    = wave * 16;

  // Register-resident W_hh column slice: 16 K-fragments (128 VGPRs).
  v16bf bfr[16];
#pragma unroll
  for (int kk = 0; kk < 16; ++kk)
    bfr[kk] = load_fragB(WhhT + (size_t)n0 * HH + kk * 32, HH);

  for (int i = threadIdx.x; i < 16 * HH; i += blockDim.x) {
    const int r = i >> 9, c = i & (HH - 1);
    Hcur[r * HPAD + c] = f2bf(state[(size_t)(bbase + r) * HH + c]);
  }
  __syncthreads();

  const float bh = Bh[n0 + n];

  for (int t = 0; t < TT; ++t) {
    v8f acc = {};
#pragma unroll
    for (int kk = 0; kk < 16; ++kk) {
      v16bf a = load_fragA(&Hcur[kk * 32], HPAD);      // ds_load_b128 x2/lane
      acc = wmma_bf16(a, bfr[kk], acc);
    }
    const float* xrow = xh + (size_t)t * BB * HH;
    if (t + 1 < TT)   // uniform branch: EXEC untouched
      __builtin_prefetch(xrow + BB * HH + (size_t)(bbase + 8 * half) * HH + n0 + n, 0, 0);
    float hv[8];
#pragma unroll
    for (int r = 0; r < 8; ++r) {
      const int gb = bbase + r + 8 * half;
      hv[r] = tanhf(acc[r] + xrow[(size_t)gb * HH + n0 + n] + bh);
    }
    __syncthreads();   // everyone done reading Hcur for this step
    unsigned short* hrow = Hseq + ((size_t)t * BB + bbase) * HH;
#pragma unroll
    for (int r = 0; r < 8; ++r) {
      const int m = r + 8 * half;
      const unsigned short u = f2bf(hv[r]);
      Hcur[m * HPAD + n0 + n]       = u;
      hrow[(size_t)m * HH + n0 + n] = u;
    }
    __syncthreads();   // Hcur fully updated before next step
  }

  for (int i = threadIdx.x; i < 16 * HH; i += blockDim.x) {
    const int r = i >> 9, c = i & (HH - 1);
    Hfinal[(size_t)(bbase + r) * HH + c] = bf2f(Hcur[r * HPAD + c]);
  }
}

// ------------------------------------------- GEMM3: out = Hseq @ W_ho + B_o
// A = Hseq [T*B][H] bf16 (row = t*B + b), B = WhoT [V][H] (N-major).
__global__ __launch_bounds__(256) void k_gemm_out(
    const unsigned short* __restrict__ Hseq,
    const unsigned short* __restrict__ WhoT,
    const float* __restrict__ Bo,
    float* __restrict__ out) {                // [B][T][V]
  const int wave = threadIdx.x >> 5;
  const int lane = threadIdx.x & 31;
  const int half = lane >> 4;
  const int n    = lane & 15;
  const int m0 = blockIdx.y * 128 + wave * 16;
  const int n0 = blockIdx.x * 64;

  v8f acc[4] = {v8f{}, v8f{}, v8f{}, v8f{}};
  const unsigned short* arow = Hseq + (size_t)m0 * HH;

#pragma unroll 2
  for (int k = 0; k < HH; k += 32) {
    __builtin_prefetch(arow + k + 128, 0, 0);
    v16bf a = load_fragA(arow + k, HH);
#pragma unroll
    for (int j = 0; j < 4; ++j) {
      v16bf b = load_fragB(WhoT + (size_t)(n0 + 16 * j) * HH + k, HH);
      acc[j] = wmma_bf16(a, b, acc[j]);
    }
  }
#pragma unroll
  for (int j = 0; j < 4; ++j) {
    const int col  = n0 + 16 * j + n;
    const float bo = Bo[col];
#pragma unroll
    for (int r = 0; r < 8; ++r) {
      const int row = m0 + r + 8 * half;     // row = t*B + b
      const int tt = row >> 5;               // B = 32
      const int bb = row & (BB - 1);
      out[(size_t)bb * TT * VV + (size_t)tt * VV + col] = acc[j][r] + bo;
    }
  }
}

// ------------------------------------------------------------------- launch
extern "C" void kernel_launch(void* const* d_in, const int* in_sizes, int n_in,
                              void* d_out, int out_size, void* d_ws, size_t ws_size,
                              hipStream_t stream) {
  const float* X   = (const float*)d_in[0];   // [B][T][V]
  const float* S0  = (const float*)d_in[1];   // [B][H]
  const float* Wxh = (const float*)d_in[2];   // [V][H]
  const float* Whh = (const float*)d_in[3];   // [H][H]
  const float* Bh  = (const float*)d_in[4];   // [H]
  const float* Who = (const float*)d_in[5];   // [H][V]
  const float* Bo  = (const float*)d_in[6];   // [V]
  float* out = (float*)d_out;                 // [B][T][V] ++ [B][H]

  char* ws = (char*)d_ws;
  // workspace layout (bytes):
  unsigned short* Xbf  = (unsigned short*)(ws);              // 128 MB  bf16 X
  float*          xh   = (float*)(ws + 134217728);           // 128 MB  fp32 xh
  unsigned short* Hseq = (unsigned short*)(ws + 268435456);  //  64 MB  bf16 H_t
  unsigned short* WxhT = (unsigned short*)(ws + 335544320);  //   1 MB
  unsigned short* WhhT = (unsigned short*)(ws + 336592896);  // 512 KB
  unsigned short* WhoT = (unsigned short*)(ws + 337117184);  //   1 MB

  // precision conversion / weight transposes (N-major for B fragments)
  k_f32_to_bf16_vec4<<<4096, 256, 0, stream>>>(X, Xbf, (long long)BB * TT * VV / 4);
  k_transpose_to_bf16<<<1024, 256, 0, stream>>>(Wxh, WxhT, VV, HH);
  k_transpose_to_bf16<<<512,  256, 0, stream>>>(Whh, WhhT, HH, HH);
  k_transpose_to_bf16<<<1024, 256, 0, stream>>>(Who, WhoT, HH, VV);

  // big parallel input projection
  dim3 g1(HH / 64, (BB * TT) / 128);
  k_gemm_xh<<<g1, 256, 0, stream>>>(Xbf, WxhT, xh);

  // serial scan: only H@W_hh on the critical path, batch-split across 2 WGPs,
  // W_hh held in VGPRs for the whole scan
  k_recurrent<<<2, 1024, 0, stream>>>(S0, xh, WhhT, Bh, Hseq,
                                      out + (size_t)BB * TT * VV);

  // big parallel output projection
  dim3 g3(VV / 64, (BB * TT) / 128);
  k_gemm_out<<<g3, 256, 0, stream>>>(Hseq, WhoT, Bo, out);
}
